// SelfAttention_86423331930123
// MI455X (gfx1250) — matrix operand order repaired and checked by reading, not compiled, not run on hardware
//
#include <hip/hip_runtime.h>

// ---------------------------------------------------------------- types ----
typedef __attribute__((ext_vector_type(16))) __bf16 v16bf;
typedef __attribute__((ext_vector_type(8)))  float  v8f;
typedef __attribute__((ext_vector_type(4)))  unsigned int u32x4;
typedef __attribute__((ext_vector_type(4)))  int v4i;

union FragBF { u32x4 u[2]; v16bf v; __bf16 e[16]; };

#define WMMA_BF16(a, b, c) \
  __builtin_amdgcn_wmma_f32_16x16x32_bf16(false, (a), false, (b), (short)0, (c), false, false)

// Problem constants (from reference): N=8, S=512, E=512, H=8, D=64
#define SEQ  512
#define EMB  512
#define HEADS 8
#define HD   64

// ----------------------------------------------- async global->LDS copy ----
__device__ __forceinline__ void async_copy_b128(const __bf16* g, __bf16* l) {
#if __has_builtin(__builtin_amdgcn_global_load_async_to_lds_b128)
  __builtin_amdgcn_global_load_async_to_lds_b128(
      (__attribute__((address_space(1))) v4i*)g,
      (__attribute__((address_space(3))) v4i*)l, 0, 0);
#else
  unsigned lofs = (unsigned)(size_t)(__attribute__((address_space(3))) __bf16*)l;
  asm volatile("global_load_async_to_lds_b128 %0, %1, off"
               :: "v"(lofs), "v"(g) : "memory");
#endif
}

__device__ __forceinline__ void wait_asynccnt0() {
#if __has_builtin(__builtin_amdgcn_s_wait_asynccnt)
  __builtin_amdgcn_s_wait_asynccnt(0);
#else
  asm volatile("s_wait_asynccnt 0x0" ::: "memory");
#endif
}

// ------------------------------------------------- fragment loaders --------
// A fragment: 16x32 bf16, row-major source, rows m0..m0+15, cols k0..k0+31.
// Layout: lane<16 holds M=lane, K=0..15; lane>=16 K=16..31 (matches B loaders).
__device__ __forceinline__ v16bf load_a_bf16(const __bf16* base, int ld,
                                             int m0, int k0, int lane) {
  int m  = m0 + (lane & 15);
  int kb = (lane & 16) ? 16 : 0;
  const __bf16* p = base + (size_t)m * ld + k0 + kb;
  FragBF f;
  f.u[0] = *(const u32x4*)(p);        // K = kb .. kb+7
  f.u[1] = *(const u32x4*)(p + 8);    // K = kb+8 .. kb+15
  return f.v;
}

// B fragment, "transposed" source: B[k][n] = M[n*ld + k]  (dot along rows of M)
__device__ __forceinline__ v16bf load_bT_bf16(const __bf16* base, int ld,
                                              int n0, int k0, int lane) {
  int n  = n0 + (lane & 15);
  int kb = (lane & 16) ? 16 : 0;
  const __bf16* p = base + (size_t)n * ld + k0 + kb;
  FragBF f;
  f.u[0] = *(const u32x4*)(p);        // K = kb .. kb+7
  f.u[1] = *(const u32x4*)(p + 8);    // K = kb+8 .. kb+15
  return f.v;
}

// B fragment, direct source: B[k][n] = M[k*ld + n] (strided along K)
__device__ __forceinline__ v16bf load_bD_bf16(const __bf16* base, int ld,
                                              int k0, int n0, int lane) {
  int n  = n0 + (lane & 15);
  int kb = (lane & 16) ? 16 : 0;
  FragBF f;
#pragma unroll
  for (int t = 0; t < 16; ++t)
    f.e[t] = base[(size_t)(k0 + kb + t) * ld + n];
  return f.v;
}

// ---------------------------------------------------------------- kernels --
__global__ __launch_bounds__(256) void cvt_f32_bf16(const float* __restrict__ in,
                                                    __bf16* __restrict__ out, int n) {
  int i = blockIdx.x * 256 + threadIdx.x;
  if (i < n) out[i] = (__bf16)in[i];
}

// C = A(MxK) * W(NxK)^T + bias ; A,W bf16, acc f32. One 16x16 tile per wave.
__global__ __launch_bounds__(256) void gemm_xwT_bf16(
    const __bf16* __restrict__ A, const __bf16* __restrict__ W,
    const float* __restrict__ bias, __bf16* __restrict__ outBf,
    float* __restrict__ outF, int M, int Nn, int K) {
  int lane = threadIdx.x & 31;
  int wave = threadIdx.x >> 5;
  int tilesN = Nn >> 4;
  int tile = blockIdx.x * 8 + wave;
  if (tile >= (M >> 4) * tilesN) return;     // wave-uniform
  int m0 = (tile / tilesN) << 4;
  int n0 = (tile % tilesN) << 4;

  v8f acc = {};
  for (int k0 = 0; k0 < K; k0 += 32) {
    __builtin_prefetch(A + (size_t)(m0 + (lane & 15)) * K + k0 + 64, 0, 0);
    v16bf a = load_a_bf16(A, K, m0, k0, lane);
    v16bf b = load_bT_bf16(W, K, n0, k0, lane);
    acc = WMMA_BF16(a, b, acc);
  }
  int n  = n0 + (lane & 15);
  int mb = m0 + ((lane & 16) ? 8 : 0);
  float bs = bias ? bias[n] : 0.0f;
#pragma unroll
  for (int j = 0; j < 8; ++j) {
    float v = acc[j] + bs;
    size_t idx = (size_t)(mb + j) * Nn + n;
    if (outF)  outF[idx]  = v;
    if (outBf) outBf[idx] = (__bf16)v;
  }
}

// Fused attention: one block = one (batch, head, 16-query-row tile).
// V panel is async-copied to LDS at entry (overlaps phases A-C).
// Phase A: E = q k^T (WMMA).  Phase B: E += diag-gather of q rel^T (WMMA).
// Phase C: masked, scaled softmax in f32.  Phase D: O = P v (WMMA from LDS).
__global__ __launch_bounds__(256) void attn_fused(
    const __bf16* __restrict__ qbf, const __bf16* __restrict__ kbf,
    const __bf16* __restrict__ vbf, const __bf16* __restrict__ relbf,
    const int* __restrict__ mask, __bf16* __restrict__ obf) {
  __shared__ float  E[16][SEQ];                    // 32 KB energies
  __shared__ __align__(16) __bf16 P[16][SEQ];      // 16 KB probabilities
  __shared__ float  O[16][HD];                     //  4 KB output accum
  __shared__ __align__(16) __bf16 Vlds[SEQ][HD];   // 64 KB V panel (async-filled)

  int lane = threadIdx.x & 31;
  int wave = threadIdx.x >> 5;
  int blk  = blockIdx.x;            // ((b*H + h)*32 + itile)
  int it   = blk & 31;
  int h    = (blk >> 5) & 7;
  int b    = blk >> 8;
  int i0   = it << 4;

  const __bf16* qrow = qbf + (size_t)b * SEQ * EMB + h * HD;  // + i*EMB
  const __bf16* krow = kbf + (size_t)b * SEQ * EMB + h * HD;
  const __bf16* vrow = vbf + (size_t)b * SEQ * EMB + h * HD;
  const __bf16* rrow = relbf + h * HD;                        // + r*EMB

  // Kick off async V panel copy: 512 rows x 64 bf16 = 4096 b128 chunks.
  // Runs on the async engines while phases A-C do WMMA + softmax.
  for (int c = threadIdx.x; c < SEQ * 8; c += 256) {
    int j = c >> 3, seg = c & 7;
    async_copy_b128(vrow + (size_t)j * EMB + seg * 8, &Vlds[j][seg * 8]);
  }

  // Q fragments for this row tile (D=64 -> two K=32 halves), reused all phases
  v16bf qa0 = load_a_bf16(qrow, EMB, i0, 0, lane);
  v16bf qa1 = load_a_bf16(qrow, EMB, i0, 32, lane);

  for (int t = threadIdx.x; t < 16 * HD; t += 256) (&O[0][0])[t] = 0.0f;

  // ---- Phase A: content energies, 4 j-tiles per wave ----
  for (int jt = wave; jt < 32; jt += 8) {
    int j0 = jt << 4;
    v8f acc = {};
    acc = WMMA_BF16(qa0, load_bT_bf16(krow, EMB, j0, 0, lane), acc);
    acc = WMMA_BF16(qa1, load_bT_bf16(krow, EMB, j0, 32, lane), acc);
    int n = lane & 15, mb = (lane & 16) ? 8 : 0;
#pragma unroll
    for (int j = 0; j < 8; ++j) E[mb + j][j0 + n] = acc[j];
  }
  __syncthreads();

  // ---- Phase B: relative bias G[i,r] = q.rel[r]; scatter to col j=r+i-512 ----
  for (int rt = wave; rt < 64; rt += 8) {
    int r0 = rt << 4;
    v8f acc = {};
    acc = WMMA_BF16(qa0, load_bT_bf16(rrow, EMB, r0, 0, lane), acc);
    acc = WMMA_BF16(qa1, load_bT_bf16(rrow, EMB, r0, 32, lane), acc);
    int n = lane & 15, mb = (lane & 16) ? 8 : 0;
#pragma unroll
    for (int j = 0; j < 8; ++j) {
      int i   = mb + j;
      int col = (r0 + n) + (i0 + i) - SEQ;   // idx = j - i + 512  =>  j = r + i - 512
      if (col >= 0 && col < SEQ) E[i][col] += acc[j];  // unique (i,col): race-free
    }
  }
  __syncthreads();

  // ---- Phase C: mask + scale + softmax (2 rows per wave, f32) ----
  const float scale = 0.125f;                 // 1/sqrt(64)
  const int* mrow = mask + (size_t)b * SEQ * SEQ;
  for (int r = wave * 2; r < wave * 2 + 2; ++r) {
    int ig = i0 + r;
    float mx = -3.4e38f;
    for (int c = lane; c < SEQ; c += 32) {
      float e = E[r][c];
      if (mrow[(size_t)ig * SEQ + c] == 0) e = -1e20f;
      e *= scale;
      E[r][c] = e;
      mx = fmaxf(mx, e);
    }
#pragma unroll
    for (int off = 16; off; off >>= 1) mx = fmaxf(mx, __shfl_xor(mx, off, 32));
    float sum = 0.0f;
    for (int c = lane; c < SEQ; c += 32) {
      float p = __expf(E[r][c] - mx);
      sum += p;
      E[r][c] = p;
    }
#pragma unroll
    for (int off = 16; off; off >>= 1) sum += __shfl_xor(sum, off, 32);
    float inv = 1.0f / sum;
    for (int c = lane; c < SEQ; c += 32) P[r][c] = (__bf16)(E[r][c] * inv);
  }

  // Drain this wave's async copies, then barrier => V fully in LDS for all.
  wait_asynccnt0();
  __syncthreads();

  // ---- Phase D: O = P(16x512) * V(512x64) from LDS; waves split (d, K/2) ----
  {
    int n0 = (wave & 3) << 4;                 // d-tile 0..3
    int kh = (wave >> 2) * 256;               // K half
    v8f acc = {};
    for (int k0 = kh; k0 < kh + 256; k0 += 32) {
      v16bf a  = load_a_bf16(&P[0][0], SEQ, 0, k0, lane);      // from LDS
      v16bf bD = load_bD_bf16(&Vlds[0][0], HD, k0, n0, lane);  // from LDS
      acc = WMMA_BF16(a, bD, acc);
    }
    int n = lane & 15, mb = (lane & 16) ? 8 : 0;
#pragma unroll
    for (int j = 0; j < 8; ++j) atomicAdd(&O[mb + j][n0 + n], acc[j]);  // ds_add_f32
  }
  __syncthreads();

  for (int t = threadIdx.x; t < 16 * HD; t += 256) {
    int i = t >> 6, d = t & 63;
    obf[((size_t)(b * SEQ + i0 + i)) * EMB + h * HD + d] = (__bf16)O[i][d];
  }
}

// ---------------------------------------------------------------- launch ---
extern "C" void kernel_launch(void* const* d_in, const int* in_sizes, int n_in,
                              void* d_out, int out_size, void* d_ws, size_t ws_size,
                              hipStream_t stream) {
  const float* values  = (const float*)d_in[0];
  const float* keys    = (const float*)d_in[1];
  const float* queries = (const float*)d_in[2];
  const int*   mask    = (const int*)d_in[3];
  const float* W_v = (const float*)d_in[4];  const float* b_v = (const float*)d_in[5];
  const float* W_k = (const float*)d_in[6];  const float* b_k = (const float*)d_in[7];
  const float* W_q = (const float*)d_in[8];  const float* b_q = (const float*)d_in[9];
  const float* W_o = (const float*)d_in[10]; const float* b_o = (const float*)d_in[11];
  const float* rel = (const float*)d_in[12];
  float* out = (float*)d_out;

  const size_t NX = (size_t)8 * SEQ * EMB;   // 2,097,152 per activation
  const size_t NW = (size_t)EMB * EMB;       //   262,144 per weight
  const size_t NR = (size_t)2 * SEQ * EMB;   //   524,288 rel table

  __bf16* ws  = (__bf16*)d_ws;
  __bf16* xv  = ws;            __bf16* xk  = xv  + NX;  __bf16* xq  = xk + NX;
  __bf16* Wvb = xq  + NX;      __bf16* Wkb = Wvb + NW;
  __bf16* Wqb = Wkb + NW;      __bf16* Wob = Wqb + NW;
  __bf16* rlb = Wob + NW;
  __bf16* qb  = rlb + NR;      __bf16* kb  = qb  + NX;  __bf16* vb = kb + NX;
  __bf16* ob  = vb  + NX;      // attention output (bf16 staging for final proj)

  auto conv = [&](const float* s, __bf16* d, size_t n) {
    cvt_f32_bf16<<<dim3((unsigned)((n + 255) / 256)), dim3(256), 0, stream>>>(s, d, (int)n);
  };
  conv(values, xv, NX);  conv(keys, xk, NX);  conv(queries, xq, NX);
  conv(W_v, Wvb, NW);    conv(W_k, Wkb, NW);  conv(W_q, Wqb, NW);
  conv(W_o, Wob, NW);    conv(rel, rlb, NR);

  // Projections: [4096,512] x [512,512]^T  -> 8192 tiles / 8 waves = 1024 blocks
  dim3 gg(1024), bb(256);
  gemm_xwT_bf16<<<gg, bb, 0, stream>>>(xv, Wvb, b_v, vb, nullptr, 4096, EMB, EMB);
  gemm_xwT_bf16<<<gg, bb, 0, stream>>>(xk, Wkb, b_k, kb, nullptr, 4096, EMB, EMB);
  gemm_xwT_bf16<<<gg, bb, 0, stream>>>(xq, Wqb, b_q, qb, nullptr, 4096, EMB, EMB);

  // Fused attention: 8 batches * 8 heads * 32 row-tiles = 2048 blocks
  attn_fused<<<dim3(2048), dim3(256), 0, stream>>>(qb, kb, vb, rlb, mask, ob);

  // Output projection -> f32 result
  gemm_xwT_bf16<<<gg, bb, 0, stream>>>(ob, Wob, b_o, nullptr, out, 4096, EMB, EMB);
}
